// ContextEncoder_30322469110470
// MI455X (gfx1250) — matrix operand order repaired and checked by reading, not compiled, not run on hardware
//
#include <hip/hip_runtime.h>
#include <cmath>

typedef __attribute__((ext_vector_type(16))) _Float16 v16h;
typedef __attribute__((ext_vector_type(8)))  float    v8f;

// ---------------------------------------------------------------------------
// Problem constants: B=128, N=128, L=128, HID=CD=128, HEADS=4, T=L-1=127
// ---------------------------------------------------------------------------
#define WHH_SW_HALFS   (384*128)      // 49152 f16 -> 98304 B
#define GATW_SW_HALFS  (512*128)      // 65536 f16 -> 131072 B
#define GATW_SW_OFF    98304
#define XH_SW_OFF      229376         // 128 batches * 16384 f16 = 4 MB

// GRU LDS layout (bytes)
#define G_WHH   0        // 98304  swizzled whh f16
#define G_HSW   98304    // 16384  swizzled h f16  (64 rows x 128)
#define G_WIH   114688   // 3072   wih f32 [384][2]
#define G_BIH   117760   // 1536
#define G_BHH   119296   // 1536
#define G_VC    120832   // 256
#define G_AC    121088   // 256
#define GRU_LDS 121344

// GAT LDS layout (bytes)
#define XLS     513                    // padded row stride (floats) for xl
#define A_XL    0                      // 128*513*4 = 262656
#define A_ASRC  262656                 // 512 f32
#define A_WLS   264704                 // 512 f32
#define A_OUT0  266752                 // 512 f32
#define A_ADST  268800                 // 4 f32
#define A_MAXH  268816
#define A_SUMH  268832
#define GAT_LDS 268848

__device__ __forceinline__ float fast_sigmoid(float x) {
  return 1.f / (1.f + __expf(-x));
}
__device__ __forceinline__ float fast_tanh(float x) {
  return 1.f - 2.f / (__expf(2.f * x) + 1.f);
}

// ---------------------------------------------------------------------------
// Prep: swizzle whh [384,128] and gat_w [512,128] (f32, row-major, row=out,
// col=in) into WMMA B-fragment order:
//   frag[(nt*4+kt)*32 + lane][e] = W[nt*16 + (lane&15)][kt*32 + 8*(lane>>4) + (e&7) + 16*(e>>3)]
// ---------------------------------------------------------------------------
__global__ __launch_bounds__(256) void prep_kernel(
    const float* __restrict__ whh, const float* __restrict__ gatw,
    _Float16* __restrict__ whh_sw, _Float16* __restrict__ gatw_sw)
{
  int g = blockIdx.x * 256 + threadIdx.x;
  if (g < WHH_SW_HALFS) {
    int e = g & 15, lane = (g >> 4) & 31, kt = (g >> 9) & 3, nt = g >> 11;
    int n = nt * 16 + (lane & 15);
    int k = kt * 32 + ((lane >> 4) << 3) + (e & 7) + ((e >> 3) << 4);
    whh_sw[g] = (_Float16)whh[n * 128 + k];
  } else {
    int i = g - WHH_SW_HALFS;
    if (i < GATW_SW_HALFS) {
      int e = i & 15, lane = (i >> 4) & 31, kt = (i >> 9) & 3, nt = i >> 11;
      int n = nt * 16 + (lane & 15);
      int k = kt * 32 + ((lane >> 4) << 3) + (e & 7) + ((e >> 3) << 4);
      gatw_sw[i] = (_Float16)gatw[n * 128 + k];
    }
  }
}

// ---------------------------------------------------------------------------
// GRU: 256 WGs x 256 threads (8 waves). WG handles 64 consecutive bn rows
// (one half-batch). Wave wv -> mt = wv>>1 (row tile), jbase = (wv&1)*4
// (column groups). Each (mt,j) pair owns r/z/n WMMA tiles + h in registers.
// Per step: 12 output tiles per wave, B frags double-buffered from LDS,
// gates computed immediately per column group; h_sw rewritten after barrier.
// ---------------------------------------------------------------------------
__global__ __launch_bounds__(256) void gru_kernel(
    const float* __restrict__ x,
    const float* __restrict__ wih, const float* __restrict__ bih,
    const float* __restrict__ bhh,
    const _Float16* __restrict__ whh_sw,
    _Float16* __restrict__ xh_sw)
{
  extern __shared__ char smem[];
  _Float16* whh_l = (_Float16*)(smem + G_WHH);
  _Float16* h_sw  = (_Float16*)(smem + G_HSW);
  float* wih_l = (float*)(smem + G_WIH);
  float* bih_l = (float*)(smem + G_BIH);
  float* bhh_l = (float*)(smem + G_BHH);
  float* vcur  = (float*)(smem + G_VC);
  float* acur  = (float*)(smem + G_AC);

  const int tid  = threadIdx.x;
  const int lane = tid & 31;
  const int wv   = tid >> 5;
  const int mt   = wv >> 1;
  const int jbase = (wv & 1) * 4;
  const int nl   = lane & 15;
  const int half = lane >> 4;

  // stage weights / biases into LDS
  {
    const uint4* src = (const uint4*)whh_sw;
    uint4* dst = (uint4*)whh_l;
    #pragma unroll 4
    for (int i = tid; i < (WHH_SW_HALFS * 2) / 16; i += 256) dst[i] = src[i];
  }
  for (int i = tid; i < 768; i += 256) wih_l[i] = wih[i];
  for (int i = tid; i < 384; i += 256) { bih_l[i] = bih[i]; bhh_l[i] = bhh[i]; }
  for (int i = tid; i < 8192; i += 256) ((unsigned*)h_sw)[i] = 0u;   // h0 = 0

  float hreg[4][8];
  #pragma unroll
  for (int jj = 0; jj < 4; ++jj)
    #pragma unroll
    for (int e = 0; e < 8; ++e) hreg[jj][e] = 0.f;

  // per-row sequence state (threads 0..63 drive one bn row each)
  const int bn = blockIdx.x * 64 + tid;
  const float2* x2 = (const float2*)x;
  float2 xprev = make_float2(0.f, 0.f);
  float vprev = 0.f;
  if (tid < 64) xprev = x2[bn * 128];

  // h_sw write coordinates (per lane, fixed across steps within each jj)
  // precompute per-jj column constants
  int c0_[4], hswbase_[4];
  #pragma unroll
  for (int jj = 0; jj < 4; ++jj) {
    int c0 = (jbase + jj) * 16 + nl;
    c0_[jj] = c0;
    int kt = c0 >> 5, k32 = c0 & 31;
    int eidx = ((k32 >> 4) << 3) | (k32 & 7);
    int lh   = (k32 >> 3) & 1;
    hswbase_[jj] = ((mt * 4 + kt) * 32 + (lh << 4)) * 16 + eidx;  // add ml*16
  }

  __syncthreads();

  for (int t = 0; t < 127; ++t) {
    // --- phase 1: speed / angle features for this step ---
    if (tid < 64) {
      float2 xc = x2[bn * 128 + t + 1];
      float dx = xc.x - xprev.x, dy = xc.y - xprev.y;
      float v  = sqrtf(dx * dx + dy * dy);
      float pv = (t == 0) ? v : vprev;
      float cs = (pv * v) / ((pv + 1e-6f) * (v + 1e-6f));
      cs = fminf(1.f, fmaxf(-1.f, cs));
      vcur[tid] = v;
      acur[tid] = acosf(cs);
      vprev = v; xprev = xc;
    }
    __syncthreads();   // S1: vcur ready; h_sw writes from prev step visible

    // --- phase 2: 12 WMMA tiles with double-buffered B frags; gate per jj ---
    v16h afr[4];
    #pragma unroll
    for (int kt = 0; kt < 4; ++kt)
      afr[kt] = *(const v16h*)(h_sw + ((mt * 4 + kt) * 32 + lane) * 16);

    auto loadB = [&](int ti, v16h dst[4]) {
      int jj = ti / 3, g = ti % 3;
      int nt = (jbase + jj) + 8 * g;
      #pragma unroll
      for (int kt = 0; kt < 4; ++kt)
        dst[kt] = *(const v16h*)(whh_l + ((nt * 4 + kt) * 32 + lane) * 16);
    };

    v16h bb[2][4];
    loadB(0, bb[0]);
    v8f accg[3];

    #pragma unroll
    for (int ti = 0; ti < 12; ++ti) {
      if (ti < 11) loadB(ti + 1, bb[(ti & 1) ^ 1]);
      v8f c = {};
      #pragma unroll
      for (int kt = 0; kt < 4; ++kt)
        c = __builtin_amdgcn_wmma_f32_16x16x32_f16(
              false, afr[kt], false, bb[ti & 1][kt], (short)0, c, false, false);
      accg[ti % 3] = c;

      if ((ti % 3) == 2) {
        const int jj = ti / 3;
        const int c0 = c0_[jj];
        float wr0 = wih_l[2*c0],       wr1 = wih_l[2*c0+1];
        float wz0 = wih_l[2*(c0+128)], wz1 = wih_l[2*(c0+128)+1];
        float wn0 = wih_l[2*(c0+256)], wn1 = wih_l[2*(c0+256)+1];
        float br = bih_l[c0],     bhr = bhh_l[c0];
        float bz = bih_l[c0+128], bhz = bhh_l[c0+128];
        float bb2 = bih_l[c0+256], bhn = bhh_l[c0+256];
        #pragma unroll
        for (int e = 0; e < 8; ++e) {
          int ml = e + 8 * half;
          float v = vcur[mt * 16 + ml], a = acur[mt * 16 + ml];
          float r  = fast_sigmoid(accg[0][e] + bhr + v*wr0 + a*wr1 + br);
          float z  = fast_sigmoid(accg[1][e] + bhz + v*wz0 + a*wz1 + bz);
          float ng = fast_tanh(v*wn0 + a*wn1 + bb2 + r * (accg[2][e] + bhn));
          hreg[jj][e] = (1.f - z) * ng + z * hreg[jj][e];
        }
      }
    }
    __syncthreads();   // S2: all h_sw reads done before rewriting it

    // --- phase 3: publish new h (f16, A-fragment layout) ---
    #pragma unroll
    for (int jj = 0; jj < 4; ++jj) {
      #pragma unroll
      for (int e = 0; e < 8; ++e) {
        int ml = e + 8 * half;
        h_sw[hswbase_[jj] + ml * 16] = (_Float16)hreg[jj][e];
      }
    }
  }
  __syncthreads();

  // --- final hidden -> global ws, already in WMMA A-fragment layout ---
  const int b     = blockIdx.x >> 1;
  const int nbase = (blockIdx.x & 1) * 64;
  #pragma unroll
  for (int jj = 0; jj < 4; ++jj) {
    int c0 = c0_[jj];
    int kt = c0 >> 5, k32 = c0 & 31;
    int eidx = ((k32 >> 4) << 3) | (k32 & 7);
    int lh   = (k32 >> 3) & 1;
    #pragma unroll
    for (int e = 0; e < 8; ++e) {
      int ml = e + 8 * half;
      int n  = nbase + mt * 16 + ml;
      int nta = n >> 4;
      xh_sw[b * 16384 + ((nta * 4 + kt) * 32 + ((n & 15) | (lh << 4))) * 16 + eidx]
          = (_Float16)hreg[jj][e];
    }
  }
}

// ---------------------------------------------------------------------------
// GAT: one WG (8 waves) per batch. xl = xh @ gat_w^T via WMMA into 256 KB LDS,
// then attention only for target node i=0 (the only row the reference keeps).
// ---------------------------------------------------------------------------
__global__ __launch_bounds__(256) void gat_kernel(
    const _Float16* __restrict__ xh_sw,
    const _Float16* __restrict__ gatw_sw,
    const float* __restrict__ att_src,
    const float* __restrict__ att_dst,
    const float* __restrict__ gbias,
    float* __restrict__ out)
{
  extern __shared__ char smem[];
  float* xl   = (float*)(smem + A_XL);     // [128][XLS]
  float* asrc = (float*)(smem + A_ASRC);   // [4*128]
  float* wls  = (float*)(smem + A_WLS);    // [4*128]
  float* out0 = (float*)(smem + A_OUT0);   // [4*128]
  float* adst = (float*)(smem + A_ADST);   // [4]
  float* maxh = (float*)(smem + A_MAXH);   // [4]
  float* sumh = (float*)(smem + A_SUMH);   // [4]

  const int tid  = threadIdx.x;
  const int lane = tid & 31;
  const int wv   = tid >> 5;
  const int nl   = lane & 15;
  const int half = lane >> 4;
  const int b    = blockIdx.x;
  const _Float16* xg = xh_sw + b * 16384;

  // xl[n, o] = sum_f xh[n,f]*gat_w[o,f] ; wave wv owns node-tile mta = wv
  v16h afr[4];
  #pragma unroll
  for (int kt = 0; kt < 4; ++kt)
    afr[kt] = *(const v16h*)(xg + ((wv * 4 + kt) * 32 + lane) * 16);

  v16h bb[2][4];
  #pragma unroll
  for (int kt = 0; kt < 4; ++kt)
    bb[0][kt] = *(const v16h*)(gatw_sw + ((kt) * 32 + lane) * 16);

  #pragma unroll
  for (int nto = 0; nto < 32; ++nto) {
    if (nto < 31) {
      #pragma unroll
      for (int kt = 0; kt < 4; ++kt)
        bb[(nto & 1) ^ 1][kt] =
            *(const v16h*)(gatw_sw + (((nto + 1) * 4 + kt) * 32 + lane) * 16);
    }
    v8f c = {};
    #pragma unroll
    for (int kt = 0; kt < 4; ++kt)
      c = __builtin_amdgcn_wmma_f32_16x16x32_f16(
            false, afr[kt], false, bb[nto & 1][kt], (short)0, c, false, false);
    #pragma unroll
    for (int e = 0; e < 8; ++e)
      xl[(wv * 16 + e + 8 * half) * XLS + nto * 16 + nl] = c[e];
  }
  __syncthreads();

  // attention logits: a_src for all j, a_dst only for node 0
  for (int p = tid; p < 512; p += 256) {
    int h = p >> 7, n = p & 127;
    float s = 0.f;
    for (int cc = 0; cc < 128; ++cc)
      s += xl[n * XLS + h * 128 + cc] * att_src[h * 128 + cc];
    asrc[p] = s;
  }
  if (tid < 4) {
    float s = 0.f;
    for (int cc = 0; cc < 128; ++cc)
      s += xl[tid * 128 + cc] * att_dst[tid * 128 + cc];   // row n=0
    adst[tid] = s;
  }
  __syncthreads();

  if (tid < 4) {
    float m = -1e30f;
    for (int j = 0; j < 128; ++j) {
      float v = asrc[tid * 128 + j] + adst[tid];
      v = (v >= 0.f) ? v : 0.2f * v;                 // leaky_relu
      m = fmaxf(m, v);
    }
    maxh[tid] = m;
  }
  __syncthreads();

  for (int p = tid; p < 512; p += 256) {
    int h = p >> 7;
    float v = asrc[p] + adst[h];
    v = (v >= 0.f) ? v : 0.2f * v;
    wls[p] = __expf(v - maxh[h]);
  }
  __syncthreads();

  if (tid < 4) {
    float s = 0.f;
    for (int j = 0; j < 128; ++j) s += wls[tid * 128 + j];
    sumh[tid] = s;
  }
  __syncthreads();

  // out0[h,c] = sum_j softmax_j * xl[j,h,c]
  for (int q = tid; q < 512; q += 256) {
    int h = q >> 7, cc = q & 127;
    float s = 0.f;
    for (int j = 0; j < 128; ++j)
      s += wls[h * 128 + j] * xl[j * XLS + h * 128 + cc];
    out0[q] = s / sumh[h];
  }
  __syncthreads();

  if (tid < 128) {
    out[b * 128 + tid] =
        0.25f * (out0[tid] + out0[128 + tid] + out0[256 + tid] + out0[384 + tid])
        + gbias[tid];
  }
}

// ---------------------------------------------------------------------------
extern "C" void kernel_launch(void* const* d_in, const int* in_sizes, int n_in,
                              void* d_out, int out_size, void* d_ws, size_t ws_size,
                              hipStream_t stream) {
  const float* x    = (const float*)d_in[0];
  const float* wih  = (const float*)d_in[1];
  const float* whh  = (const float*)d_in[2];
  const float* bihp = (const float*)d_in[3];
  const float* bhhp = (const float*)d_in[4];
  const float* gatw = (const float*)d_in[5];
  const float* asw  = (const float*)d_in[6];
  const float* adw  = (const float*)d_in[7];
  const float* gb   = (const float*)d_in[8];
  float* out = (float*)d_out;

  char* ws = (char*)d_ws;
  _Float16* whh_sw  = (_Float16*)(ws);
  _Float16* gatw_sw = (_Float16*)(ws + GATW_SW_OFF);
  _Float16* xh_sw   = (_Float16*)(ws + XH_SW_OFF);

  prep_kernel<<<448, 256, 0, stream>>>(whh, gatw, whh_sw, gatw_sw);
  gru_kernel<<<256, 256, GRU_LDS, stream>>>(x, wih, bihp, bhhp, whh_sw, xh_sw);
  gat_kernel<<<128, 256, GAT_LDS, stream>>>(xh_sw, gatw_sw, asw, adw, gb, out);
}
